// PSMMix3dEmbedding_87385404605033
// MI455X (gfx1250) — matrix operand order repaired and checked by reading, not compiled, not run on hardware
//
#include <hip/hip_runtime.h>

// ---------------------------------------------------------------------------
// MI455X (gfx1250, wave32) implementation of PSMMix3dEmbedding.
// B=4, L=2048, D=1024, K=128.
//
// Roofline: outputs + adj ~117 MB -> ~5 us @ 23.3 TB/s. GEMMs = 6.4 GFLOP:
// with f16 WMMA (16x16x32, f32 acc) compute is ~1-2 us -> memory bound.
// GEMM1 B-operands are DMA'd into LDS by the Tensor Data Mover (TENSORcnt,
// double buffered) so WMMAs never wait on global latency; the B-operand
// ds_loads are software-pipelined one feature tile ahead so WMMA execution
// covers LDS latency.
// ---------------------------------------------------------------------------

typedef __attribute__((ext_vector_type(16))) _Float16 v16h;
typedef __attribute__((ext_vector_type(8)))  _Float16 v8h;
typedef __attribute__((ext_vector_type(8)))  float    v8f;
typedef unsigned int v4u  __attribute__((ext_vector_type(4)));
typedef int          v8i32 __attribute__((ext_vector_type(8)));
typedef int          v4i32 __attribute__((ext_vector_type(4)));

#define BB 4
#define LL 2048
#define DD 1024
#define KK 128
// 128^-0.5
#define SCALE 0.08838834764831845f
// finfo(f32).min * SCALE  (masked score, matches reference dist*scaling)
#define NEGS  (-3.007714e37f)

#if __has_builtin(__builtin_amdgcn_tensor_load_to_lds) && \
    __has_builtin(__builtin_amdgcn_s_wait_tensorcnt)
#define USE_TDM 1
#else
#define USE_TDM 0
#endif

#if USE_TDM
// ---------------------------------------------------------------------------
// TDM descriptor for one pemb tile: f16 2-D tensor [128 rows][2048 cols],
// row stride 2048 elems, tile = 32 (dim0/keys) x 128 (dim1/features),
// loaded contiguously into LDS at lds_off (128 rows x 64 B = 8 KB).
// ---------------------------------------------------------------------------
__device__ inline void tdm_load_tile(const void* gsrc, unsigned lds_off) {
  unsigned long long ga = (unsigned long long)(uintptr_t)gsrc;
  v4u g0;
  g0[0] = 1u;                                   // count=1, user mode
  g0[1] = lds_off;                              // lds_addr (bytes)
  g0[2] = (unsigned)(ga & 0xffffffffu);         // global_addr[31:0]
  g0[3] = (unsigned)((ga >> 32) & 0x01ffffffu)  // global_addr[56:32]
          | (2u << 30);                         // type = 2 ("image")
  v8i32 g1;
  g1[0] = (int)(1u << 16);       // workgroup_mask=0, data_size=1 (2 bytes)
  g1[1] = (int)(2048u << 16);    // tensor_dim0[15:0] = 2048 at bits[31:16]
  g1[2] = (int)(128u << 16);     // tensor_dim0[31:16]=0 | tensor_dim1[15:0]=128
  g1[3] = (int)(32u << 16);      // tensor_dim1[31:16]=0 | tile_dim0=32
  g1[4] = 128;                   // tile_dim1=128, tile_dim2=0
  g1[5] = 2048;                  // tensor_dim0_stride[31:0] = 2048
  g1[6] = 0;                     // stride0 hi | stride1 lo
  g1[7] = 0;                     // stride1 hi
  v4i32 z4 = {0, 0, 0, 0};
#if __clang_major__ >= 23
  v8i32 z8 = {0, 0, 0, 0, 0, 0, 0, 0};
  __builtin_amdgcn_tensor_load_to_lds(g0, g1, z4, z4, z8, 0);
#else
  __builtin_amdgcn_tensor_load_to_lds(g0, g1, z4, z4, 0);
#endif
}
#endif  // USE_TDM

// ---------------------------------------------------------------------------
// 1) time-embed gather + mask bytes.  One block per (b,l) row.
//    maskbyte: bit0 = padding, bit1 = molecule.
// ---------------------------------------------------------------------------
__global__ __launch_bounds__(256)
void embed_time_kernel(const int* __restrict__ token_id,
                       const unsigned char* __restrict__ is_periodic,
                       const int* __restrict__ time_step,
                       const unsigned char* __restrict__ clean_mask,
                       const float* __restrict__ time_table,
                       float* __restrict__ out_mask,
                       float* __restrict__ out_te,
                       unsigned char* __restrict__ maskbyte) {
  int row = blockIdx.x;                 // 0..B*L-1
  int b = row >> 11;
  int t = time_step[row];
  bool clean = clean_mask[row] != 0;
  const float* src = time_table + (size_t)t * DD;
  float* dst = out_te + (size_t)row * DD;
  for (int d = threadIdx.x; d < DD; d += 256)
    dst[d] = clean ? 0.0f : src[d];
  if (threadIdx.x == 0) {
    int tok = token_id[row];
    bool pad = (tok == 0);
    bool mol = (tok <= 129) && (is_periodic[b] == 0);
    maskbyte[row] = (unsigned char)((pad ? 1 : 0) | (mol ? 2 : 0));
    out_mask[row] = pad ? 1.0f : 0.0f;
  }
}

// ---------------------------------------------------------------------------
// 2) pemb = (pad ? 0 : pos @ W_pos^T), stored TRANSPOSED as f16:
//    pembT[b][k][l]  so GEMM1 B-operands are contiguous per lane.
// ---------------------------------------------------------------------------
__global__ __launch_bounds__(128)
void pemb_kernel(const int* __restrict__ token_id,
                 const float* __restrict__ pos,
                 const float* __restrict__ W_pos,
                 _Float16* __restrict__ pembT) {
  __shared__ float sp[32 * 3];
  __shared__ unsigned char spad[32];
  int blk = blockIdx.x;                 // B * (L/32)
  int b  = blk >> 6;
  int l0 = (blk & 63) * 32;
  int tid = threadIdx.x;
  if (tid < 96) sp[tid] = pos[((size_t)b * LL + l0) * 3 + tid];
  if (tid < 32) spad[tid] = (token_id[b * LL + l0 + tid] == 0) ? 1 : 0;
  __syncthreads();
  float wx = W_pos[tid * 3 + 0];
  float wy = W_pos[tid * 3 + 1];
  float wz = W_pos[tid * 3 + 2];
  _Float16 vals[32];
#pragma unroll
  for (int l = 0; l < 32; ++l) {
    float d = sp[l * 3] * wx + sp[l * 3 + 1] * wy + sp[l * 3 + 2] * wz;
    vals[l] = (_Float16)(spad[l] ? 0.0f : d);
  }
  _Float16* dst = pembT + ((size_t)b * KK + tid) * LL + l0;
#pragma unroll
  for (int c = 0; c < 4; ++c) {
    v8h v;
#pragma unroll
    for (int e = 0; e < 8; ++e) v[e] = vals[c * 8 + e];
    *(v8h*)(dst + c * 8) = v;
  }
}

// ---------------------------------------------------------------------------
// 3) W_feat f32[D][K] -> f16 (row-major, K contiguous -> GEMM2 B-operand).
// ---------------------------------------------------------------------------
__global__ __launch_bounds__(256)
void wfeat_kernel(const float* __restrict__ W_feat,
                  _Float16* __restrict__ wf16) {
  int i = blockIdx.x * 256 + threadIdx.x;   // grid covers D*K exactly
  wf16[i] = (_Float16)W_feat[i];
}

// ---------------------------------------------------------------------------
// 4) softmax row stats: m_i = max_j s_ij, invZ_i = 1/sum_j exp(s_ij - m_i).
//    One wave (32 lanes) per row, online max/sum + shuffle reduction.
// ---------------------------------------------------------------------------
__global__ __launch_bounds__(256)
void stats_kernel(const float* __restrict__ pos,
                  const unsigned char* __restrict__ adj,
                  const unsigned char* __restrict__ maskbyte,
                  float* __restrict__ rowmax,
                  float* __restrict__ rowinvz) {
  int lane = threadIdx.x & 31;
  int row  = blockIdx.x * 8 + (threadIdx.x >> 5);
  int b = row >> 11;
  int i = row & 2047;
  unsigned char qmb = maskbyte[row];
  bool padq = qmb & 1;
  bool molq = (qmb >> 1) & 1;
  const float* posb = pos + (size_t)b * LL * 3;
  float qx = posb[i * 3 + 0], qy = posb[i * 3 + 1], qz = posb[i * 3 + 2];
  const unsigned char* adjrow = adj + (size_t)row * LL;
  const unsigned char* mbb = maskbyte + (size_t)b * LL;
  float m = -3.4e38f, Z = 0.0f;
  for (int j = lane; j < LL; j += 32) {
    bool padk = mbb[j] & 1;
    bool ab = adjrow[j] != 0;
    float dx = qx - posb[j * 3 + 0];
    float dy = qy - posb[j * 3 + 1];
    float dz = qz - posb[j * 3 + 2];
    float r = sqrtf(dx * dx + dy * dy + dz * dz);
    bool neg = padq || padk || (molq && !ab);
    float s = neg ? NEGS : (SCALE / (r + 1.0f));
    float mn = fmaxf(m, s);
    Z = Z * __expf(m - mn) + __expf(s - mn);
    m = mn;
  }
#pragma unroll
  for (int off = 16; off > 0; off >>= 1) {
    float mo = __shfl_xor(m, off, 32);
    float Zo = __shfl_xor(Z, off, 32);
    float mn = fmaxf(m, mo);
    Z = Z * __expf(m - mn) + Zo * __expf(mo - mn);
    m = mn;
  }
  if (lane == 0) { rowmax[row] = m; rowinvz[row] = 1.0f / Z; }
}

// ---------------------------------------------------------------------------
// 5) GEMM1: attnout[b,l,k] = softmax(scores) @ pemb.
//    Block = 4 waves, one 16-row query tile; each wave covers 1/4 of the key
//    range (16 tiles of 32 keys). Per wave the 8 KB pemb B-tile is staged in
//    LDS by the Tensor Data Mover (double buffered, s_wait_tensorcnt), the
//    probability tile is built directly in the WMMA A-operand lane layout,
//    and 8x v_wmma_f32_16x16x32_f16 accumulate per tile with the B ds_loads
//    pipelined one tile ahead. Wave partials are reduced through LDS.
// ---------------------------------------------------------------------------
__global__ __launch_bounds__(128)
void gemm1_kernel(const float* __restrict__ pos,
                  const unsigned char* __restrict__ adj,
                  const unsigned char* __restrict__ maskbyte,
                  const float* __restrict__ rowmax,
                  const float* __restrict__ rowinvz,
                  const _Float16* __restrict__ pembT,
                  _Float16* __restrict__ attnout) {
  __shared__ char smem[65536];   // [4 waves][2 bufs][8 KB]  /  32 KB reduce
  int b  = blockIdx.x >> 7;
  int q0 = (blockIdx.x & 127) << 4;
  int tid  = threadIdx.x;
  int w    = tid >> 5;
  int lane = tid & 31;
  int mrow = lane & 15;
  int kh   = lane >> 4;
  int qrow = q0 + mrow;
  size_t rowidx = (size_t)b * LL + qrow;
  unsigned char qmb = maskbyte[rowidx];
  bool padq = qmb & 1;
  bool molq = (qmb >> 1) & 1;
  const float* posb = pos + (size_t)b * LL * 3;
  float qx = posb[qrow * 3 + 0], qy = posb[qrow * 3 + 1], qz = posb[qrow * 3 + 2];
  float rmax = rowmax[rowidx];
  float rinv = rowinvz[rowidx];

  v8f acc[8] = {};
  const unsigned char* adjbase = adj + rowidx * LL;
  const unsigned char* mbb = maskbyte + (size_t)b * LL;
  const _Float16* pb = pembT + (size_t)b * KK * LL;

  char* stage = smem + w * 16384;
  const int kt0 = w * 16;                 // this wave's first key tile
#if USE_TDM
  unsigned lds0 = ((unsigned)(uintptr_t)(void*)smem) + (unsigned)(w * 16384);
  tdm_load_tile(pb + (size_t)kt0 * 32, lds0);      // prologue: buffer 0
#endif

  for (int t = 0; t < 16; ++t) {
    int kt = kt0 + t;
    int base = kt * 32;
#if USE_TDM
    if (t + 1 < 16) {                      // prefetch next tile, other buffer
      tdm_load_tile(pb + (size_t)(base + 32), lds0 + (unsigned)(((t + 1) & 1) * 8192));
      __builtin_amdgcn_s_wait_tensorcnt(1);
    } else {
      __builtin_amdgcn_s_wait_tensorcnt(0);
    }
#else
    {  // same-wave cooperative copy fallback: 8 KB = 32 lanes x 16 x 16 B
      const char* gsrc = (const char*)pb + (size_t)base * 2;
      char* dstb = stage + (t & 1) * 8192;
#pragma unroll
      for (int c = 0; c < 16; ++c) {
        int u = lane + c * 32;             // 16-byte unit
        int row = u >> 2, ch = u & 3;
        *(v4u*)(dstb + u * 16) =
            *(const v4u*)(gsrc + (size_t)row * 4096 + ch * 16);
      }
    }
#endif
    __builtin_prefetch(adjbase + base + 32, 0, 3);

    // each lane loads one key's position; share via ds_bpermute
    int keyj = base + lane;
    float kxl = posb[keyj * 3 + 0];
    float kyl = posb[keyj * 3 + 1];
    float kzl = posb[keyj * 3 + 2];
    unsigned long long a0 = *(const unsigned long long*)(adjbase + base + kh * 16);
    unsigned long long a1 = *(const unsigned long long*)(adjbase + base + 8 + kh * 16);
    unsigned long long w0 = *(const unsigned long long*)(mbb + base + kh * 16);
    unsigned long long w1 = *(const unsigned long long*)(mbb + base + 8 + kh * 16);

    v16h A;
#pragma unroll
    for (int v = 0; v < 8; ++v) {
      int kk0 = 2 * v + kh * 16; // K slot in tile
      unsigned long long aw = (v < 4) ? a0 : a1;
      unsigned long long ww = (v < 4) ? w0 : w1;
      int bidx = 2 * (v & 3);
#pragma unroll
      for (int e = 0; e < 2; ++e) {
        int kkt = kk0 + e;
        bool ab = ((aw >> ((bidx + e) * 8)) & 0xffULL) != 0;
        bool pk = ((ww >> ((bidx + e) * 8)) & 0x1ULL) != 0;
        float kx = __shfl(kxl, kkt, 32);
        float ky = __shfl(kyl, kkt, 32);
        float kz = __shfl(kzl, kkt, 32);
        float dx = qx - kx, dy = qy - ky, dz = qz - kz;
        float r = sqrtf(dx * dx + dy * dy + dz * dz);
        bool neg = padq || pk || (molq && !ab);
        float s = neg ? NEGS : (SCALE / (r + 1.0f));
        float p = __expf(s - rmax) * rinv;     // normalized prob
        A[2 * v + e] = (_Float16)p;
      }
    }
    // B operands from LDS: row = feature (f*16 + n), 16 consecutive keys.
    // Rotating 1-deep pipeline: issue loads for tile f+1 before WMMA f so
    // WMMA execution hides the LDS round-trip (no per-f s_wait_dscnt stall).
    const char* bufc = stage + (t & 1) * 8192;
    v16h Bcur = *(const v16h*)(bufc + (size_t)mrow * 64 + kh * 32);
#pragma unroll
    for (int f = 0; f < 8; ++f) {
      v16h Bnext = Bcur;
      if (f < 7)
        Bnext = *(const v16h*)(bufc + ((f + 1) * 16 + mrow) * 64 + kh * 32);
      acc[f] = __builtin_amdgcn_wmma_f32_16x16x32_f16(
          false, A, false, Bcur, (short)0, acc[f], false, false);
      Bcur = Bnext;
    }
  }

  // ---- cross-wave reduction of the 4 key-range partials ----
  __syncthreads();                         // staging buffers now dead
  float* red = (float*)smem;               // [4][16 rows][128 cols] f32
#pragma unroll
  for (int f = 0; f < 8; ++f)
#pragma unroll
    for (int r = 0; r < 8; ++r)
      red[(w * 16 + r + 8 * kh) * 128 + f * 16 + mrow] = acc[f][r];
  __syncthreads();
  int rr = tid >> 3;                       // 0..15 : output row
  int c0 = (tid & 7) * 16;                 // 16-col chunk
  v8h o0v, o1v;
#pragma unroll
  for (int e = 0; e < 16; ++e) {
    int c = c0 + e;
    float s = red[(0 * 16 + rr) * 128 + c] + red[(1 * 16 + rr) * 128 + c] +
              red[(2 * 16 + rr) * 128 + c] + red[(3 * 16 + rr) * 128 + c];
    if (e < 8) o0v[e] = (_Float16)s; else o1v[e - 8] = (_Float16)s;
  }
  _Float16* orow = attnout + ((size_t)b * LL + q0 + rr) * KK + c0;
  *(v8h*)(orow) = o0v;
  *(v8h*)(orow + 8) = o1v;
}

// ---------------------------------------------------------------------------
// 6) GEMM2: pfe = attnout @ W_feat^T, fused epilogue:
//    out3 = where(pad,0,pfe) + time_embed ; out0 = embed_table[token] + out3
//    One wave per (b, 16-row qtile, 128-col dtile). K loop = 128/32 = 4;
//    all 8 B operands loaded as one clause before the 8 WMMAs.
// ---------------------------------------------------------------------------
__global__ __launch_bounds__(32)
void gemm2_kernel(const _Float16* __restrict__ attnout,
                  const _Float16* __restrict__ wf16,
                  const unsigned char* __restrict__ maskbyte,
                  const int* __restrict__ token_id,
                  const float* __restrict__ embed_table,
                  const float* __restrict__ out_te,
                  float* __restrict__ out_x,
                  float* __restrict__ out_pt) {
  int blk = blockIdx.x;                 // B * 128 * 8
  int dt = blk & 7;
  int qt = (blk >> 3) & 127;
  int b  = blk >> 10;
  int q0 = qt << 4;
  int d0 = dt << 7;
  int lane = threadIdx.x;
  int n  = lane & 15;
  int kh = lane >> 4;

  v8f acc[8] = {};
  const _Float16* arow = attnout + ((size_t)b * LL + q0 + n) * KK; // A row M=n
#pragma unroll
  for (int kt = 0; kt < 4; ++kt) {
    int kb = kt * 32;
    v8h lo = *(const v8h*)(arow + kb + kh * 16);
    v8h hi = *(const v8h*)(arow + kb + 8 + kh * 16);
    v16h A;
#pragma unroll
    for (int e = 0; e < 8; ++e) { A[e] = lo[e]; A[8 + e] = hi[e]; }
    v16h Bt[8];
#pragma unroll
    for (int f = 0; f < 8; ++f)
      Bt[f] = *(const v16h*)(wf16 + (size_t)(d0 + f * 16 + n) * KK + kb + kh * 16);
#pragma unroll
    for (int f = 0; f < 8; ++f)
      acc[f] = __builtin_amdgcn_wmma_f32_16x16x32_f16(
          false, A, false, Bt[f], (short)0, acc[f], false, false);
  }
  unsigned long long pads =
      *(const unsigned long long*)(maskbyte + (size_t)b * LL + q0 + kh * 8);
#pragma unroll
  for (int r = 0; r < 8; ++r) {
    int orow = q0 + r + 8 * kh;
    bool pad = ((pads >> (8 * r)) & 1) != 0;
    int tok = token_id[b * LL + orow];
    size_t rofs = ((size_t)b * LL + orow) * DD;
    const float* erow = embed_table + (size_t)tok * DD;
#pragma unroll
    for (int f = 0; f < 8; ++f) {
      int col = d0 + f * 16 + n;
      float pfe = pad ? 0.0f : acc[f][r];
      float o3 = pfe + out_te[rofs + col];
      out_pt[rofs + col] = o3;
      out_x[rofs + col] = erow[col] + o3;
    }
  }
}

// ---------------------------------------------------------------------------
extern "C" void kernel_launch(void* const* d_in, const int* in_sizes, int n_in,
                              void* d_out, int out_size, void* d_ws,
                              size_t ws_size, hipStream_t stream) {
  (void)in_sizes; (void)n_in; (void)out_size; (void)ws_size;
  const int*           token_id    = (const int*)d_in[0];
  const unsigned char* is_periodic = (const unsigned char*)d_in[1];
  const float*         pos         = (const float*)d_in[2];
  const unsigned char* adj         = (const unsigned char*)d_in[3];
  const int*           time_step   = (const int*)d_in[4];
  const unsigned char* clean_mask  = (const unsigned char*)d_in[5];
  const float*         embed_table = (const float*)d_in[6];
  const float*         time_table  = (const float*)d_in[7];
  const float*         W_pos       = (const float*)d_in[8];
  const float*         W_feat      = (const float*)d_in[9];

  const size_t NBL = (size_t)BB * LL;
  float* out      = (float*)d_out;
  float* out_x    = out;                    // [B,L,D]
  float* out_mask = out_x + NBL * DD;       // [B,L]
  float* out_te   = out_mask + NBL;         // [B,L,D]
  float* out_pt   = out_te + NBL * DD;      // [B,L,D]

  char* ws = (char*)d_ws;
  _Float16*      pembT    = (_Float16*)(ws + 0);            // 2 MB
  _Float16*      attnout  = (_Float16*)(ws + 2097152);      // 2 MB
  _Float16*      wf16     = (_Float16*)(ws + 4194304);      // 256 KB
  unsigned char* maskbyte = (unsigned char*)(ws + 4456448); // 8 KB
  float*         rowmax   = (float*)(ws + 4464640);         // 32 KB
  float*         rowinvz  = (float*)(ws + 4497408);         // 32 KB

  embed_time_kernel<<<(int)NBL, 256, 0, stream>>>(
      token_id, is_periodic, time_step, clean_mask, time_table,
      out_mask, out_te, maskbyte);
  pemb_kernel<<<BB * (LL / 32), 128, 0, stream>>>(token_id, pos, W_pos, pembT);
  wfeat_kernel<<<(DD * KK) / 256, 256, 0, stream>>>(W_feat, wf16);
  stats_kernel<<<(int)(NBL / 8), 256, 0, stream>>>(
      pos, adj, maskbyte, rowmax, rowinvz);
  gemm1_kernel<<<BB * (LL / 16), 128, 0, stream>>>(
      pos, adj, maskbyte, rowmax, rowinvz, pembT, attnout);
  gemm2_kernel<<<BB * (LL / 16) * (DD / 128), 32, 0, stream>>>(
      attnout, wf16, maskbyte, token_id, embed_table, out_te, out_x, out_pt);
}
